// FullAttention_30021821399886
// MI455X (gfx1250) — matrix operand order repaired and checked by reading, not compiled, not run on hardware
//
#include <hip/hip_runtime.h>

typedef __attribute__((ext_vector_type(16))) __bf16 v16bf;
typedef __attribute__((ext_vector_type(8)))  __bf16 v8bf;
typedef __attribute__((ext_vector_type(8)))  float  v8f;

#define BB 4
#define LL 2048
#define HH 16
#define EE 64
#define MT 128            // query rows per block (8 waves x 16)
#define KT 64             // key rows per tile
#define NWAVE 8
#define RS 72             // LDS row stride in bf16 elems (64 + 8 pad, 144B = 9*16B)
#define SCALE_LOG2E 0.1803368787f   // (1/sqrt(64)) * log2(e), folded into Q

__device__ __forceinline__ __bf16 f2bf(float f) {
  unsigned u = __builtin_bit_cast(unsigned, f);
  u += 0x7FFFu + ((u >> 16) & 1u);           // round-to-nearest-even
  unsigned short hb = (unsigned short)(u >> 16);
  return __builtin_bit_cast(__bf16, hb);
}

__device__ __forceinline__ v16bf ld_frag16(const __bf16* p) {
  v8bf a = *(const v8bf*)p;
  v8bf b = *(const v8bf*)(p + 8);
  return __builtin_shufflevector(a, b, 0,1,2,3,4,5,6,7,8,9,10,11,12,13,14,15);
}

__global__ __launch_bounds__(256)
void fa_diagband_kernel(const float* __restrict__ Q, const float* __restrict__ K,
                        const float* __restrict__ V, float* __restrict__ O) {
  __shared__ __bf16 sK[KT * RS];             // [key row][feat]
  __shared__ __bf16 sV[EE * RS];             // transposed: [feat][key]
  __shared__ __bf16 sP[NWAVE * 16 * RS];     // per-wave P tile, [row][key]

  const int tid  = threadIdx.x;
  const int wave = tid >> 5;
  const int lane = tid & 31;
  const int l16  = lane & 15;
  const int lh   = lane >> 4;                // lane half (0/1)

  const int bid = blockIdx.x;
  const int qt  = bid & 15;                  // L/MT = 16 query tiles
  const int h   = (bid >> 4) & 15;
  const int b   = bid >> 8;
  const int i0  = qt * MT;
  const int off = (1 << (h >> 2)) - 1;       // {0,1,3,7}

  // ---- load Q fragments (A-layout, bf16), softmax scale folded in ----
  const int qi = i0 + wave * 16 + l16;
  const float* qrow = Q + (((size_t)b * LL + qi) * HH + h) * EE;
  v16bf qf[2];
#pragma unroll
  for (int f = 0; f < 2; ++f) {
    const int e0 = 32 * f + 16 * lh;         // elems 0..15 -> K = e0..e0+15
#pragma unroll
    for (int t = 0; t < 8; ++t) qf[f][t]     = f2bf(qrow[e0 + t] * SCALE_LOG2E);
#pragma unroll
    for (int t = 0; t < 8; ++t) qf[f][8 + t] = f2bf(qrow[e0 + 8 + t] * SCALE_LOG2E);
  }

  v8f acc[4];
#pragma unroll
  for (int nt = 0; nt < 4; ++nt)
#pragma unroll
    for (int v = 0; v < 8; ++v) acc[nt][v] = 0.f;

  float mrow[8], lrow[8];                    // lrow is lane-partial (1 col per lane)
#pragma unroll
  for (int v = 0; v < 8; ++v) { mrow[v] = -__builtin_inff(); lrow[v] = 0.f; }

  // cooperative-load geometry: each thread owns 16 consecutive feats of one row
  const int cr = tid >> 2;                   // tile row 0..63
  const int cc = (tid & 3) << 4;             // feat start 0/16/32/48
  const size_t kvBase = ((size_t)b * LL * HH + (size_t)cr * HH + h) * EE + cc;

  // ---- register staging: raw f32 for the tile currently being fetched ----
  float kst[16], vst[16];
  {
    const float* kr = K + kvBase + (size_t)i0 * HH * EE;
    const float* vr = V + kvBase + (size_t)i0 * HH * EE;
#pragma unroll
    for (int t = 0; t < 16; ++t) kst[t] = kr[t];
#pragma unroll
    for (int t = 0; t < 16; ++t) vst[t] = vr[t];
  }

  // valid key tiles are exactly j0 = i0 .. L-1 (anti-causal band, off < 64,
  // and the always-visible last column lives in tile j0 = L-KT >= i0)
  for (int j0 = i0; j0 < LL; j0 += KT) {
    __syncthreads();                         // all waves done reading prev tile
    {
#pragma unroll
      for (int t = 0; t < 16; ++t) sK[cr * RS + cc + t]   = f2bf(kst[t]);
#pragma unroll
      for (int t = 0; t < 16; ++t) sV[(cc + t) * RS + cr] = f2bf(vst[t]);  // transpose
    }
    __syncthreads();

    // prefetch next tile into registers: no consumer until next iteration's
    // store phase, so these global loads overlap the WMMA/softmax below
    if (j0 + KT < LL) {
      const float* kr = K + kvBase + (size_t)(j0 + KT) * HH * EE;
      const float* vr = V + kvBase + (size_t)(j0 + KT) * HH * EE;
#pragma unroll
      for (int t = 0; t < 16; ++t) kst[t] = kr[t];
#pragma unroll
      for (int t = 0; t < 16; ++t) vst[t] = vr[t];
    }
    if (j0 + 2 * KT < LL) {                  // warm L2 two tiles ahead
      __builtin_prefetch(K + kvBase + (size_t)(j0 + 2 * KT) * HH * EE, 0, 1);
      __builtin_prefetch(V + kvBase + (size_t)(j0 + 2 * KT) * HH * EE, 0, 1);
    }

    // ---- S = Q * K^T (4 tiles of 16x16, K-dim 64 = 2 x wmma k32) ----
    v8f s[4];
#pragma unroll
    for (int nt = 0; nt < 4; ++nt) {
      v8f c;
#pragma unroll
      for (int v = 0; v < 8; ++v) c[v] = 0.f;
#pragma unroll
      for (int kf = 0; kf < 2; ++kf) {
        v16bf bf = ld_frag16(&sK[(nt * 16 + l16) * RS + kf * 32 + 16 * lh]);
        c = __builtin_amdgcn_wmma_f32_16x16x32_bf16(false, qf[kf], false, bf,
                                                    (short)0, c, false, false);
      }
      s[nt] = c;                             // already in exp2 domain (scale in Q)
    }

    // ---- boundary masking: keep (j > i+off) || (j == L-1) ----
    if (j0 <= i0 + MT - 1 + off) {
#pragma unroll
      for (int nt = 0; nt < 4; ++nt) {
        const int j = j0 + nt * 16 + l16;
#pragma unroll
        for (int v = 0; v < 8; ++v) {
          const int i = i0 + wave * 16 + v + 8 * lh;
          const bool vis = (j > i + off) || (j == LL - 1);
          if (!vis) s[nt][v] = -__builtin_inff();
        }
      }
    }

    // ---- online softmax (rows live across 16-lane halves) ----
    float mnew[8];
#pragma unroll
    for (int v = 0; v < 8; ++v) {
      float mx = s[0][v];
#pragma unroll
      for (int nt = 1; nt < 4; ++nt) mx = fmaxf(mx, s[nt][v]);
#pragma unroll
      for (int d = 1; d < 16; d <<= 1) mx = fmaxf(mx, __shfl_xor(mx, d, 32));
      mnew[v] = fmaxf(mrow[v], mx);
    }
#pragma unroll
    for (int v = 0; v < 8; ++v) {
      const float a = (mnew[v] == -__builtin_inff()) ? 0.f : exp2f(mrow[v] - mnew[v]);
      lrow[v] *= a;
#pragma unroll
      for (int nt = 0; nt < 4; ++nt) acc[nt][v] *= a;
      mrow[v] = mnew[v];
    }
    __bf16* pw = &sP[(wave * 16) * RS];
#pragma unroll
    for (int nt = 0; nt < 4; ++nt) {
#pragma unroll
      for (int v = 0; v < 8; ++v) {
        const float p = (mnew[v] == -__builtin_inff()) ? 0.f : exp2f(s[nt][v] - mnew[v]);
        lrow[v] += p;
        pw[(v + 8 * lh) * RS + nt * 16 + l16] = f2bf(p);   // C-layout -> LDS
      }
    }

    // ---- O += P * V (A-frags of P from LDS, B-frags of V^T from LDS) ----
#pragma unroll
    for (int nt = 0; nt < 4; ++nt) {
      v8f c = acc[nt];
#pragma unroll
      for (int kf = 0; kf < 2; ++kf) {
        const __bf16* pa = &sP[(wave * 16 + l16) * RS + kf * 32 + 16 * lh];
        v8bf a0 = *(const v8bf*)pa;
        v8bf a1 = *(const v8bf*)(pa + 8);
        v16bf af = __builtin_shufflevector(a0, a1,
                     0,1,2,3,4,5,6,7,8,9,10,11,12,13,14,15);
        v16bf bf = ld_frag16(&sV[(nt * 16 + l16) * RS + kf * 32 + 16 * lh]);
        c = __builtin_amdgcn_wmma_f32_16x16x32_bf16(false, af, false, bf,
                                                    (short)0, c, false, false);
      }
      acc[nt] = c;
    }
  }

  // ---- epilogue: reduce row sums across 16 lanes, scale, store ----
#pragma unroll
  for (int v = 0; v < 8; ++v) {
    float l = lrow[v];
#pragma unroll
    for (int d = 1; d < 16; d <<= 1) l += __shfl_xor(l, d, 32);
    const float inv = (l > 0.f) ? 1.f / l : 0.f;
    const int i = i0 + wave * 16 + v + 8 * lh;
    float* orow = O + (((size_t)b * LL + i) * HH + h) * EE;
#pragma unroll
    for (int nt = 0; nt < 4; ++nt)
      orow[nt * 16 + l16] = acc[nt][v] * inv;
  }
}

extern "C" void kernel_launch(void* const* d_in, const int* in_sizes, int n_in,
                              void* d_out, int out_size, void* d_ws, size_t ws_size,
                              hipStream_t stream) {
  (void)in_sizes; (void)n_in; (void)out_size; (void)d_ws; (void)ws_size;
  const float* Q = (const float*)d_in[0];
  const float* K = (const float*)d_in[1];
  const float* V = (const float*)d_in[2];
  float* O = (float*)d_out;
  dim3 grid(BB * HH * (LL / MT));   // 1024 blocks: (b, h, qtile)
  fa_diagband_kernel<<<grid, dim3(256), 0, stream>>>(Q, K, V, O);
}
